// SpatialConv_4002909520607
// MI455X (gfx1250) — matrix operand order repaired and checked by reading, not compiled, not run on hardware
//
#include <hip/hip_runtime.h>

typedef __attribute__((ext_vector_type(16))) __bf16 v16bf;
typedef __attribute__((ext_vector_type(8)))  float  v8f;
typedef __attribute__((ext_vector_type(4)))  unsigned int u32x4;

#define BB 16
#define CC 128
#define HH 72
#define WW 200
#define KT 9
#define HWSZ (HH * WW)          // 14400
#define CHW  (CC * HWSZ)        // 1843200
#define WPK1 (KT * CC * CC)     // elements per bf16 weight set
#define LPMAX (WW + 8)          // padded line length (max)

__device__ __forceinline__ unsigned short f2bf(float f) {
    unsigned int u = __float_as_uint(f);
    u += 0x7FFFu + ((u >> 16) & 1u);      // round-to-nearest-even
    return (unsigned short)(u >> 16);
}

union ABPack { u32x4 u[2]; v16bf v; };

// Repack 4 weight tensors [co][ci][k] fp32 -> bf16 [dir][k][co][ci]
__global__ void pack_weights(const float* __restrict__ w0, const float* __restrict__ w1,
                             const float* __restrict__ w2, const float* __restrict__ w3,
                             unsigned short* __restrict__ wpk) {
    int i = blockIdx.x * 256 + threadIdx.x;
    const int total = 4 * WPK1;
    if (i >= total) return;
    int ci = i & 127;
    int co = (i >> 7) & 127;
    int k  = (i >> 14) % KT;
    int d  = (i >> 14) / KT;
    const float* w = (d == 0) ? w0 : (d == 1) ? w1 : (d == 2) ? w2 : w3;
    wpk[i] = f2bf(w[(co * CC + ci) * KT + k]);
}

// Convert one scan line of fp32 out -> padded bf16 buffer [b][l+4][c];
// writes zero halos (4 columns each side) into BOTH ping-pong buffers.
__global__ void cvt_line(const float* __restrict__ out,
                         unsigned short* __restrict__ buf0,
                         unsigned short* __restrict__ buf1,
                         int lineOff, int lineStride, int L) {
    int Lp = L + 8;
    int i = blockIdx.x * 256 + threadIdx.x;
    if (i >= BB * Lp * CC) return;
    int c  = i & 127;
    int lp = (i >> 7) % Lp;
    int b  = i / (CC * Lp);
    int l  = lp - 4;
    if (l < 0 || l >= L) {
        buf0[i] = 0;
        buf1[i] = 0;          // halos persist: step kernels only rewrite the interior
    } else {
        size_t idx = (size_t)b * CHW + (size_t)c * HWSZ + (size_t)lineOff + (size_t)l * lineStride;
        buf0[i] = f2bf(out[idx]);
    }
}

// One recurrence step: out[line] += relu(conv1d(prev line) + bias); emit bf16 copy of new line.
__global__ __launch_bounds__(256) void scnn_step(
    float* __restrict__ out,
    const unsigned short* __restrict__ wpk,     // [k][co][ci] bf16 for this direction
    const float* __restrict__ bias,
    const unsigned short* __restrict__ prevbuf, // padded [b][l+4][ci] bf16
    unsigned short* __restrict__ curbuf,        // padded [b][l+4][ci] bf16
    int lineOff, int lineStride, int L, int lTiles)
{
    int lane = threadIdx.x & 31;
    int wav  = threadIdx.x >> 5;
    int t = blockIdx.x * 8 + wav;
    int tilesPerB = 8 * lTiles;
    if (t >= BB * tilesPerB) return;          // wave-uniform exit

    int bIdx = t / tilesPerB;
    int rem  = t % tilesPerB;
    int coT  = rem / lTiles;                  // lT varies fastest: neighbor waves share A tiles
    int lT   = rem % lTiles;
    int coBase = coT * 16;
    int lBase  = lT * 16;

    int Lp   = L + 8;
    int half = lane >> 4;                     // 0: lanes 0-15, 1: lanes 16-31
    int n    = lane & 15;                     // N column (and A row m) within tile

    v8f acc = {};

    // A: lane owns row co = coBase+n; per-lane K groups {half*8..+7, 16+half*8..+7}
    const unsigned short* wb = wpk + (size_t)(coBase + n) * CC + half * 8;
    const unsigned short* pb = prevbuf + (size_t)bIdx * Lp * CC + half * 8;

    for (int k = 0; k < KT; ++k) {
        // padded index: (lBase+n-4+k) + 4 = lBase+n+k; clamp overhang of partial tiles
        int lp = lBase + n + k;
        lp = (lp > Lp - 1) ? (Lp - 1) : lp;
        const unsigned short* bp = pb + (size_t)lp * CC;
        const unsigned short* ap = wb + (size_t)k * (CC * CC);
#pragma unroll
        for (int cb = 0; cb < 4; ++cb) {
            u32x4 a0 = *(const u32x4*)(ap + cb * 32);
            u32x4 a1 = *(const u32x4*)(ap + cb * 32 + 16);
            u32x4 b0 = *(const u32x4*)(bp + cb * 32);
            u32x4 b1 = *(const u32x4*)(bp + cb * 32 + 16);
            ABPack A;  A.u[0] = a0; A.u[1] = a1;
            ABPack Bv; Bv.u[0] = b0; Bv.u[1] = b1;
            acc = __builtin_amdgcn_wmma_f32_16x16x32_bf16(
                false, A.v, false, Bv.v, (short)0, acc, false, false);
        }
    }

    // Epilogue: C layout -> lane holds N = n, M = r + 8*half for r = 0..7
    int l = lBase + n;
    if (l < L) {
        int coB = coBase + half * 8;
        size_t base = (size_t)bIdx * CHW + (size_t)lineOff + (size_t)l * lineStride;
        unsigned short pk[8];
#pragma unroll
        for (int r = 0; r < 8; ++r) {
            int co = coB + r;
            size_t idx = base + (size_t)co * HWSZ;
            float old  = out[idx];
            float conv = acc[r] + bias[co];
            float v    = old + (conv > 0.f ? conv : 0.f);
            out[idx] = v;
            pk[r] = f2bf(v);
        }
        // 8 consecutive channels -> one 16B store into the padded bf16 line buffer
        *(u32x4*)(curbuf + ((size_t)bIdx * Lp + (l + 4)) * CC + coB) = *(const u32x4*)pk;
    }
}

extern "C" void kernel_launch(void* const* d_in, const int* in_sizes, int n_in,
                              void* d_out, int out_size, void* d_ws, size_t ws_size,
                              hipStream_t stream) {
    const float* x   = (const float*)d_in[0];
    const float* w_d = (const float*)d_in[1]; const float* b_d = (const float*)d_in[2];
    const float* w_u = (const float*)d_in[3]; const float* b_u = (const float*)d_in[4];
    const float* w_r = (const float*)d_in[5]; const float* b_r = (const float*)d_in[6];
    const float* w_l = (const float*)d_in[7]; const float* b_l = (const float*)d_in[8];
    float* out = (float*)d_out;

    unsigned short* wpk  = (unsigned short*)d_ws;          // 4*WPK1 bf16
    unsigned short* buf0 = wpk + 4 * WPK1;                 // padded [b][l+4][c]
    unsigned short* buf1 = buf0 + (size_t)BB * LPMAX * CC;
    unsigned short* bufs[2] = { buf0, buf1 };

    // out = x
    hipMemcpyAsync(out, x, sizeof(float) * (size_t)BB * CHW, hipMemcpyDeviceToDevice, stream);

    // bf16 weight repack [dir][k][co][ci]
    {
        int total = 4 * WPK1;
        pack_weights<<<(total + 255) / 256, 256, 0, stream>>>(w_d, w_u, w_r, w_l, wpk);
    }

    auto runPass = [&](const unsigned short* wp, const float* bias,
                       int L, int nLines, int scanStride, int lineStride, bool fwd) {
        int start  = fwd ? 0 : nLines - 1;
        int dir    = fwd ? 1 : -1;
        int nsteps = fwd ? nLines - 1 : nLines - 2;   // bwd pass leaves line 0 untouched
        int lTiles = (L + 15) / 16;
        int cvtN   = BB * (L + 8) * CC;
        cvt_line<<<(cvtN + 255) / 256, 256, 0, stream>>>(out, bufs[0], bufs[1],
                                                         start * scanStride, lineStride, L);
        int blocks = BB * lTiles;                     // 16*8*lTiles tiles / 8 waves per block
        for (int i = 0; i < nsteps; ++i) {
            int line = start + (i + 1) * dir;
            scnn_step<<<blocks, 256, 0, stream>>>(out, wp, bias,
                                                  bufs[i & 1], bufs[(i + 1) & 1],
                                                  line * scanStride, lineStride, L, lTiles);
        }
    };

    // down / up: scan over H (scanStride=W), conv along W (lineStride=1, L=W)
    runPass(wpk + 0 * WPK1, b_d, WW, HH, WW, 1, true);
    runPass(wpk + 1 * WPK1, b_u, WW, HH, WW, 1, false);
    // right / left: scan over W (scanStride=1), conv along H (lineStride=W, L=H)
    runPass(wpk + 2 * WPK1, b_r, HH, WW, 1, WW, true);
    runPass(wpk + 3 * WPK1, b_l, HH, WW, 1, WW, false);
}
